// QuantumAttention_50122268344529
// MI455X (gfx1250) — compile-verified, matrix operand order checked by reading, not run on hardware
//
#include <hip/hip_runtime.h>
#include <hip/hip_bf16.h>

#define BB 2
#define SS 2048
#define NN 512
#define VOCAB 50257
#define ROWS (BB * SS)          // 4096

typedef _Float16 v16h __attribute__((ext_vector_type(16)));
typedef _Float16 v8h  __attribute__((ext_vector_type(8)));
typedef float    v8f  __attribute__((ext_vector_type(8)));

// ---------------- small utility kernels ----------------

__global__ void k_embed(const int* __restrict__ x, const float* __restrict__ r0,
                        float* __restrict__ rv, int n) {
    int i = blockIdx.x * blockDim.x + threadIdx.x;
    if (i < n) {
        float v = r0[x[i]];
        rv[i] = v / fmaxf(fabsf(v), 1e-12f);
    }
}

__global__ void k_cvt_f16(const float* __restrict__ src, _Float16* __restrict__ dst, int n) {
    int i = blockIdx.x * blockDim.x + threadIdx.x;
    if (i < n) dst[i] = (_Float16)src[i];
}

__device__ __forceinline__ float block_reduce(float v, float* sbuf) {
    int tid = threadIdx.x;
    sbuf[tid] = v;
    __syncthreads();
    for (int off = 128; off > 0; off >>= 1) {
        if (tid < off) sbuf[tid] += sbuf[tid + off];
        __syncthreads();
    }
    float r = sbuf[0];
    __syncthreads();
    return r;
}

// Pass 1 of the Heun integrator: row stats on r_embed, thresholded causal
// matvec -> k1, target = r + k1.  One block (256 thr) per (b, i) row.
__global__ __launch_bounds__(256) void k_pass1(const float* __restrict__ rv,
                                               const float* __restrict__ dt,
                                               float* __restrict__ k1,
                                               float* __restrict__ target) {
    __shared__ float sbuf[256];
    int row = blockIdx.x;
    int b = row / SS, i = row % SS;
    const float* r = rv + b * SS;
    float ri = r[i];

    float sum = 0.f, sumsq = 0.f;
    for (int k = threadIdx.x; k < SS; k += 256) {
        float d = fabsf(ri - r[k]);
        sum += d; sumsq += d * d;
    }
    sum   = block_reduce(sum, sbuf);
    sumsq = block_reduce(sumsq, sbuf);
    float mean = sum / (float)SS;
    float var  = fmaxf((sumsq - (float)SS * mean * mean) / (float)(SS - 1), 0.f);
    float thr  = fmaxf(mean + 2.f * sqrtf(var), 1e-12f);

    float acc = 0.f;
    for (int k = threadIdx.x; k < i; k += 256) {      // strictly causal (tril k=-1)
        float d = fabsf(ri - r[k]);
        if (d <= thr) acc += (1.f - d / thr) * r[k];
    }
    acc = block_reduce(acc, sbuf);
    if (threadIdx.x == 0) {
        float dtb = fminf(fmaxf(dt[b], 0.f), 1.f);
        float k1v = fminf(fmaxf(dtb * acc, -1.f), 1.f);
        k1[row] = k1v;
        target[row] = ri + k1v;
    }
}

// Pass 2: stats on target, emit full score_target row (output), k2, r_new,
// diff_coef.  One block per (b, i) row.
__global__ __launch_bounds__(256) void k_pass2(const float* __restrict__ tg,
                                               const float* __restrict__ rv,
                                               const float* __restrict__ k1,
                                               const float* __restrict__ dt,
                                               float* __restrict__ score_out,
                                               float* __restrict__ r_new,
                                               float* __restrict__ diff_coef) {
    __shared__ float sbuf[256];
    int row = blockIdx.x;
    int b = row / SS, i = row % SS;
    const float* t = tg + b * SS;
    float ti = t[i];

    float sum = 0.f, sumsq = 0.f;
    for (int k = threadIdx.x; k < SS; k += 256) {
        float d = fabsf(ti - t[k]);
        sum += d; sumsq += d * d;
    }
    sum   = block_reduce(sum, sbuf);
    sumsq = block_reduce(sumsq, sbuf);
    float mean = sum / (float)SS;
    float var  = fmaxf((sumsq - (float)SS * mean * mean) / (float)(SS - 1), 0.f);
    float thr  = fmaxf(mean + 2.f * sqrtf(var), 1e-12f);

    float acc2 = 0.f, ssum = 0.f;
    float* srow = score_out + (size_t)row * SS;
    for (int k = threadIdx.x; k < SS; k += 256) {
        float s = 0.f;
        if (k < i) {
            float d = fabsf(ti - t[k]);
            if (d <= thr) s = 1.f - d / thr;
        }
        srow[k] = s;
        acc2 += s * t[k];
        ssum += s;
    }
    acc2 = block_reduce(acc2, sbuf);
    ssum = block_reduce(ssum, sbuf);
    if (threadIdx.x == 0) {
        float dtb = fminf(fmaxf(dt[b], 0.f), 1.f);
        float k2  = fminf(fmaxf(dtb * acc2, -1.f), 1.f);
        float rn  = rv[row] + 0.5f * (k1[row] + k2);
        r_new[row] = rn / fmaxf(fabsf(rn), 1e-12f);
        diff_coef[row] = fminf(fmaxf(ssum / (float)SS, 0.f), 1.f);
    }
}

// Build final_rep in f16: [ROWS, NN] = [r_new | clip(diff*w_diff + b_diff)]
__global__ void k_rep(const float* __restrict__ r_new, const float* __restrict__ diff,
                      const float* __restrict__ w_diff, const float* __restrict__ b_diff,
                      _Float16* __restrict__ A) {
    int idx = blockIdx.x * blockDim.x + threadIdx.x;
    if (idx >= ROWS * NN) return;
    int row = idx >> 9;      // / 512
    int n   = idx & (NN - 1);
    float v;
    if (n == 0) v = r_new[row];
    else {
        int j = n - 1;
        v = fminf(fmaxf(diff[row] * w_diff[j] + b_diff[j], -1.f), 1.f);
    }
    A[idx] = (_Float16)v;
}

__global__ void k_meandt(const float* __restrict__ dt, float* __restrict__ out) {
    if (threadIdx.x == 0) {
        float a = fminf(fmaxf(dt[0], 0.f), 1.f);
        float b = fminf(fmaxf(dt[1], 0.f), 1.f);
        out[0] = 0.5f * (a + b);
    }
}

// ---------------- WMMA GEMM: logits = clip(A @ W^T + bias, +-2) ----------------
// A: [ROWS, NN] f16, W: [VOCAB, NN] f16 (B column n == W row n), C: [ROWS, VOCAB] f32
// Each wave: two 16-row M tiles x four 16-col N tiles (B fragments reused),
// K looped by 32.  Out-of-range B rows are clamped (results discarded at store),
// so the hot loop has no EXEC masking and no zero-fills.
__device__ __forceinline__ v16h ldfrag(const _Float16* p) {
    v8h lo = *(const v8h*)p;
    v8h hi = *(const v8h*)(p + 16);
    return __builtin_shufflevector(lo, hi, 0,1,2,3,4,5,6,7,8,9,10,11,12,13,14,15);
}

__global__ __launch_bounds__(256) void k_gemm(const _Float16* __restrict__ A,
                                              const _Float16* __restrict__ W,
                                              const float* __restrict__ bias,
                                              float* __restrict__ C) {
    int wave = threadIdx.x >> 5;
    int lane = threadIdx.x & 31;
    int wid  = blockIdx.x * 8 + wave;
    int mt = wid & 127;          // 128 groups of 32 rows (4096 / 32)
    int ng = wid >> 7;           // N group of 64 columns
    int m0 = mt << 5;
    int n0 = ng << 6;
    int lr = lane & 15;
    int s  = (lane < 16) ? 0 : 8;   // K sub-offset per 16-bit A/B fragment layout

    const _Float16* Ap0 = A + (size_t)(m0 + lr) * NN + s;
    const _Float16* Ap1 = A + (size_t)(m0 + 16 + lr) * NN + s;
    const _Float16* Bp[4];
#pragma unroll
    for (int t = 0; t < 4; t++) {
        int brow = n0 + t * 16 + lr;
        if (brow > VOCAB - 1) brow = VOCAB - 1;   // clamp: garbage cols never stored
        Bp[t] = W + (size_t)brow * NN + s;
    }

    v8f acc0[4] = {};
    v8f acc1[4] = {};
#pragma unroll 2
    for (int k0 = 0; k0 < NN; k0 += 32) {
        v16h a0 = ldfrag(Ap0 + k0);
        v16h a1 = ldfrag(Ap1 + k0);
#pragma unroll
        for (int t = 0; t < 4; t++) {
            v16h bm = ldfrag(Bp[t] + k0);
            acc0[t] = __builtin_amdgcn_wmma_f32_16x16x32_f16(
                        false, a0, false, bm, (short)0, acc0[t], false, false);
            acc1[t] = __builtin_amdgcn_wmma_f32_16x16x32_f16(
                        false, a1, false, bm, (short)0, acc1[t], false, false);
        }
    }

    int mbase0 = m0 + ((lane < 16) ? 0 : 8);
    int mbase1 = mbase0 + 16;
#pragma unroll
    for (int t = 0; t < 4; t++) {
        int col = n0 + t * 16 + lr;
        if (col >= VOCAB) continue;
        float bs = bias[col];
#pragma unroll
        for (int r = 0; r < 8; r++) {
            float v0 = fminf(fmaxf(acc0[t][r] + bs, -2.f), 2.f);
            float v1 = fminf(fmaxf(acc1[t][r] + bs, -2.f), 2.f);
            C[(size_t)(mbase0 + r) * VOCAB + col] = v0;
            C[(size_t)(mbase1 + r) * VOCAB + col] = v1;
        }
    }
}

// ---------------- launcher ----------------

extern "C" void kernel_launch(void* const* d_in, const int* in_sizes, int n_in,
                              void* d_out, int out_size, void* d_ws, size_t ws_size,
                              hipStream_t stream) {
    const int*   x      = (const int*)d_in[0];
    const float* dt     = (const float*)d_in[1];
    const float* r0     = (const float*)d_in[2];
    const float* w_diff = (const float*)d_in[3];
    const float* b_diff = (const float*)d_in[4];
    const float* w_out  = (const float*)d_in[5];
    const float* b_out  = (const float*)d_in[6];

    float* logits    = (float*)d_out;                        // [B,S,VOCAB]
    float* score_out = logits + (size_t)ROWS * VOCAB;        // [B,S,S]
    float* mean_out  = score_out + (size_t)ROWS * SS;        // scalar

    float* rv     = (float*)d_ws;        // ROWS
    float* k1     = rv + ROWS;
    float* target = k1 + ROWS;
    float* r_new  = target + ROWS;
    float* diff   = r_new + ROWS;
    _Float16* Ah  = (_Float16*)(diff + ROWS);                // ROWS*NN halves (16B aligned)
    _Float16* Wh  = Ah + (size_t)ROWS * NN;                  // VOCAB*NN halves

    k_embed<<<(ROWS + 255) / 256, 256, 0, stream>>>(x, r0, rv, ROWS);
    k_cvt_f16<<<(VOCAB * NN + 255) / 256, 256, 0, stream>>>(w_out, Wh, VOCAB * NN);
    k_pass1<<<ROWS, 256, 0, stream>>>(rv, dt, k1, target);
    k_pass2<<<ROWS, 256, 0, stream>>>(target, rv, k1, dt, score_out, r_new, diff);
    k_rep<<<(ROWS * NN) / 256, 256, 0, stream>>>(r_new, diff, w_diff, b_diff, Ah);

    int ngroups = (VOCAB + 63) / 64;          // 786
    int nblocks = (128 * ngroups) / 8;        // 12576 blocks x 8 waves
    k_gemm<<<nblocks, 256, 0, stream>>>(Ah, Wh, b_out, logits);

    k_meandt<<<1, 32, 0, stream>>>(dt, mean_out);
}
